// Cell_40939628265850
// MI455X (gfx1250) — matrix operand order repaired and checked by reading, not compiled, not run on hardware
//
#include <hip/hip_runtime.h>

typedef __attribute__((ext_vector_type(16))) __bf16 v16bf;
typedef __attribute__((ext_vector_type(8)))  __bf16 v8bf;
typedef __attribute__((ext_vector_type(8)))  float  v8f;
typedef __attribute__((ext_vector_type(4)))  float  f32x4;

#define NROWS 16384
#define DDIM  1024
#define CCONST 8.0f
#define LDS_RS 1032                      // padded row stride (elements): conflict-free
#define LDS_MAT_ELEMS (32 * LDS_RS)      // one 32-row weight strip
#define LDS_BYTES (2 * LDS_MAT_ELEMS * 2)  // Wa + Wx strips, bf16

// ---------------------------------------------------------------------------
// Helpers
// ---------------------------------------------------------------------------
__device__ __forceinline__ float sigm(float x) {
    return 1.0f / (1.0f + __expf(-x));
}

// Async global -> LDS copy of 16 bytes (ASYNCcnt-tracked, CDNA5 path)
__device__ __forceinline__ void async_ld16(unsigned lds_off, unsigned long long gaddr) {
    asm volatile("global_load_async_to_lds_b128 %0, %1, off"
                 :: "v"(lds_off), "v"(gaddr)
                 : "memory");
}
__device__ __forceinline__ void wait_async0() {
    asm volatile("s_wait_asynccnt 0x0" ::: "memory");
}

// B-fragment: 16 contiguous bf16 at p (lane = column n, K = base..base+15)
__device__ __forceinline__ v16bf ldfrag_b(const __bf16* p) {
    v8bf lo = *(const v8bf*)p;
    v8bf hi = *(const v8bf*)(p + 8);
    return __builtin_shufflevector(lo, hi, 0,1,2,3,4,5,6,7,8,9,10,11,12,13,14,15);
}
// A-fragment: elements 0..7 = p[0..7], elements 8..15 = p[16..23]
__device__ __forceinline__ v16bf ldfrag_a(const __bf16* p) {
    v8bf lo = *(const v8bf*)p;
    v8bf hi = *(const v8bf*)(p + 16);
    return __builtin_shufflevector(lo, hi, 0,1,2,3,4,5,6,7,8,9,10,11,12,13,14,15);
}
// f32 -> bf16 fragment loaders (fallback path)
__device__ __forceinline__ v16bf cvt16(f32x4 v0, f32x4 v1, f32x4 v2, f32x4 v3) {
    v16bf r;
#pragma unroll
    for (int i = 0; i < 4; ++i) {
        r[i] = (__bf16)v0[i]; r[4+i] = (__bf16)v1[i];
        r[8+i] = (__bf16)v2[i]; r[12+i] = (__bf16)v3[i];
    }
    return r;
}
__device__ __forceinline__ v16bf ldfrag_b_f32(const float* __restrict__ p) {
    const f32x4* q = (const f32x4*)p;
    return cvt16(q[0], q[1], q[2], q[3]);
}
__device__ __forceinline__ v16bf ldfrag_a_f32(const float* __restrict__ p) {
    const f32x4* q = (const f32x4*)p;
    return cvt16(q[0], q[1], q[4], q[5]);
}

// ---------------------------------------------------------------------------
// Kernel 0: fp32 -> bf16 conversion (8 elements / thread)
// ---------------------------------------------------------------------------
__global__ __launch_bounds__(256) void cvt_bf16_kernel(
    const float* __restrict__ src, unsigned short* __restrict__ dst, int n)
{
    const int i = (blockIdx.x * blockDim.x + threadIdx.x) * 8;
    if (i >= n) return;
    const f32x4* q = (const f32x4*)(src + i);
    f32x4 a = q[0], b = q[1];
    v8bf r;
#pragma unroll
    for (int j = 0; j < 4; ++j) { r[j] = (__bf16)a[j]; r[4+j] = (__bf16)b[j]; }
    *(v8bf*)(dst + i) = r;
}

// ---------------------------------------------------------------------------
// Shared epilogue (MT = number of 16-row tiles per wave)
// ---------------------------------------------------------------------------
template<int MT>
__device__ __forceinline__ void epilogue_t(
    v8f acc[2][MT][2], int m0, int n0, int l16, int hf,
    const float* __restrict__ xt, const float* __restrict__ hprev,
    const float* __restrict__ ba, const float* __restrict__ bx,
    const float* __restrict__ Lambda, float* __restrict__ out)
{
#pragma unroll
    for (int nt = 0; nt < 2; ++nt) {
        const int nn    = n0 + nt * 16 + l16;
        const float ban = ba[nn];
        const float bxn = bx[nn];
        const float lam = Lambda[nn];
        const float sp   = fmaxf(lam, 0.0f) + log1pf(__expf(-fabsf(lam)));
        const float ncsp = -CCONST * sp;
#pragma unroll
        for (int mt = 0; mt < MT; ++mt) {
#pragma unroll
            for (int j = 0; j < 8; ++j) {
                const int m = m0 + mt * 16 + hf * 8 + j;
                const size_t idx = (size_t)m * DDIM + nn;
                const float xv = xt[idx];
                const float hv = hprev[idx];
                const float rt = sigm(acc[0][mt][nt][j] + ban);
                const float it = sigm(acc[1][mt][nt][j] + bxn);
                const float at = __expf(ncsp * rt);
                const float w  = sqrtf(fmaxf(1.0f - at * at, 0.0f));
                out[idx] = at * hv + w * (it * xv);
            }
        }
    }
}

// ---------------------------------------------------------------------------
// Kernel 1: fused dual-GEMM + epilogue. bf16 operands; weights staged in LDS
// via async global->LDS copies; 64x32 output tile per wave.
// Block = 8 waves, all sharing one 32-column weight strip (Wa+Wx, full K).
// ---------------------------------------------------------------------------
__global__ __launch_bounds__(256) void minlstm_bf16_kernel(
    const unsigned short* __restrict__ xb_,   // bf16 xt [NROWS][DDIM]
    const unsigned short* __restrict__ wab_,  // bf16 Wa [DDIM][DDIM]
    const unsigned short* __restrict__ wxb_,  // bf16 Wx [DDIM][DDIM]
    const float* __restrict__ xt,
    const float* __restrict__ hprev,
    const float* __restrict__ ba,
    const float* __restrict__ bx,
    const float* __restrict__ Lambda,
    float* __restrict__ out)
{
    extern __shared__ __align__(16) char smem[];
    __bf16* lds = (__bf16*)smem;                       // [0]: Wa strip, [LDS_MAT_ELEMS]: Wx strip

    const __bf16* xb = (const __bf16*)xb_;

    const int tid   = threadIdx.x;
    const int lane  = tid & 31;
    const int l16   = lane & 15;
    const int hf    = lane >> 4;
    const int wslot = tid >> 5;

    const int nstrip = blockIdx.x & 31;                // 32 col strips
    const int mstrip = (blockIdx.x >> 5) * 8 + wslot;  // 256 row strips (64 rows each)
    const int m0 = mstrip * 64;
    const int n0 = nstrip * 32;

    // ---- Stage Wa/Wx strips (rows n0..n0+31, all K) into padded LDS -------
    // One matrix strip = 32 rows * 128 chunks of 16B.  8192 chunks total.
    {
        const unsigned lds0 = (unsigned)(size_t)lds;   // LDS base byte offset
        for (int c = tid; c < 8192; c += 256) {
            const int mat   = c >> 12;                 // 0 = Wa, 1 = Wx
            const int cc    = c & 4095;
            const int row   = cc >> 7;                 // 0..31
            const int chunk = cc & 127;                // 16B chunks along K
            const unsigned short* src = (mat ? wxb_ : wab_)
                + (size_t)(n0 + row) * DDIM + chunk * 8;
            const unsigned loff = lds0 + (unsigned)(mat * LDS_MAT_ELEMS + row * LDS_RS) * 2
                                + chunk * 16;
            async_ld16(loff, (unsigned long long)(size_t)src);
        }
        wait_async0();
    }
    __syncthreads();

    // ---- Accumulators: [gemm][mtile(4)][ntile(2)] = 128 VGPRs ------------
    v8f acc[2][4][2];
#pragma unroll
    for (int g = 0; g < 2; ++g)
#pragma unroll
        for (int mt = 0; mt < 4; ++mt)
#pragma unroll
            for (int nt = 0; nt < 2; ++nt)
                acc[g][mt][nt] = (v8f){};

    const __bf16* xr[4] = { xb + (size_t)(m0      + l16) * DDIM,
                            xb + (size_t)(m0 + 16 + l16) * DDIM,
                            xb + (size_t)(m0 + 32 + l16) * DDIM,
                            xb + (size_t)(m0 + 48 + l16) * DDIM };
    const __bf16* lwa[2] = { lds + (l16)      * LDS_RS,
                             lds + (16 + l16) * LDS_RS };
    const __bf16* lwx[2] = { lds + LDS_MAT_ELEMS + (l16)      * LDS_RS,
                             lds + LDS_MAT_ELEMS + (16 + l16) * LDS_RS };

    for (int kb = 0; kb < DDIM; kb += 32) {
        const int ka  = kb + hf * 8;    // A: K {ka..ka+7, ka+16..ka+23}
        const int kbb = kb + hf * 8;    // B: K {kbb..kbb+7, kbb+16..kbb+23}

        v16bf aF[4], bFa[2], bFx[2];
#pragma unroll
        for (int mt = 0; mt < 4; ++mt) aF[mt]  = ldfrag_a(xr[mt] + ka);
#pragma unroll
        for (int nt = 0; nt < 2; ++nt) bFa[nt] = ldfrag_a(lwa[nt] + kbb);
#pragma unroll
        for (int nt = 0; nt < 2; ++nt) bFx[nt] = ldfrag_a(lwx[nt] + kbb);

#pragma unroll
        for (int mt = 0; mt < 4; ++mt)
#pragma unroll
            for (int nt = 0; nt < 2; ++nt) {
                acc[0][mt][nt] = __builtin_amdgcn_wmma_f32_16x16x32_bf16(
                    false, aF[mt], false, bFa[nt], (short)0, acc[0][mt][nt], false, false);
                acc[1][mt][nt] = __builtin_amdgcn_wmma_f32_16x16x32_bf16(
                    false, aF[mt], false, bFx[nt], (short)0, acc[1][mt][nt], false, false);
            }
    }

    epilogue_t<4>(acc, m0, n0, l16, hf, xt, hprev, ba, bx, Lambda, out);
}

// ---------------------------------------------------------------------------
// Kernel 2: fallback (no workspace) — fused GEMM converting f32 inline, 32x32
// ---------------------------------------------------------------------------
__global__ __launch_bounds__(256) void minlstm_f32src_kernel(
    const float* __restrict__ xt,
    const float* __restrict__ hprev,
    const float* __restrict__ Wa,
    const float* __restrict__ Wx,
    const float* __restrict__ ba,
    const float* __restrict__ bx,
    const float* __restrict__ Lambda,
    float* __restrict__ out)
{
    const int lane = threadIdx.x & 31;
    const int l16  = lane & 15;
    const int hf   = lane >> 4;
    const int wid  = blockIdx.x * (blockDim.x >> 5) + (threadIdx.x >> 5);
    const int m0 = (wid >> 5) * 32;
    const int n0 = (wid & 31) * 32;

    v8f acc[2][2][2];
#pragma unroll
    for (int g = 0; g < 2; ++g)
#pragma unroll
        for (int mt = 0; mt < 2; ++mt)
#pragma unroll
            for (int nt = 0; nt < 2; ++nt)
                acc[g][mt][nt] = (v8f){};

    const float* xr[2]  = { xt + (size_t)(m0 + l16) * DDIM,
                            xt + (size_t)(m0 + 16 + l16) * DDIM };
    const float* war[2] = { Wa + (size_t)(n0 + l16) * DDIM,
                            Wa + (size_t)(n0 + 16 + l16) * DDIM };
    const float* wxr[2] = { Wx + (size_t)(n0 + l16) * DDIM,
                            Wx + (size_t)(n0 + 16 + l16) * DDIM };

    for (int kb = 0; kb < DDIM; kb += 32) {
        const int ka  = kb + hf * 8;
        const int kbb = kb + hf * 8;

        v16bf aF[2], bFa[2], bFx[2];
#pragma unroll
        for (int mt = 0; mt < 2; ++mt) aF[mt]  = ldfrag_a_f32(xr[mt] + ka);
#pragma unroll
        for (int nt = 0; nt < 2; ++nt) bFa[nt] = ldfrag_a_f32(war[nt] + kbb);
#pragma unroll
        for (int nt = 0; nt < 2; ++nt) bFx[nt] = ldfrag_a_f32(wxr[nt] + kbb);

#pragma unroll
        for (int mt = 0; mt < 2; ++mt)
#pragma unroll
            for (int nt = 0; nt < 2; ++nt) {
                acc[0][mt][nt] = __builtin_amdgcn_wmma_f32_16x16x32_bf16(
                    false, aF[mt], false, bFa[nt], (short)0, acc[0][mt][nt], false, false);
                acc[1][mt][nt] = __builtin_amdgcn_wmma_f32_16x16x32_bf16(
                    false, aF[mt], false, bFx[nt], (short)0, acc[1][mt][nt], false, false);
            }
    }

    epilogue_t<2>(acc, m0, n0, l16, hf, xt, hprev, ba, bx, Lambda, out);
}

// ---------------------------------------------------------------------------
// Launch
// ---------------------------------------------------------------------------
extern "C" void kernel_launch(void* const* d_in, const int* in_sizes, int n_in,
                              void* d_out, int out_size, void* d_ws, size_t ws_size,
                              hipStream_t stream) {
    const float* xt     = (const float*)d_in[0];
    const float* hprev  = (const float*)d_in[1];
    const float* Wa     = (const float*)d_in[2];
    const float* Wx     = (const float*)d_in[3];
    const float* ba     = (const float*)d_in[4];
    const float* bx     = (const float*)d_in[5];
    const float* Lambda = (const float*)d_in[6];
    float* out = (float*)d_out;

    const size_t n_xt = (size_t)NROWS * DDIM;
    const size_t n_w  = (size_t)DDIM * DDIM;
    const size_t need = (n_xt + 2 * n_w) * sizeof(unsigned short);

    if (ws_size >= need) {
        unsigned short* xb  = (unsigned short*)d_ws;
        unsigned short* wab = xb + n_xt;
        unsigned short* wxb = wab + n_w;

        cvt_bf16_kernel<<<(int)(n_xt / 8 / 256), 256, 0, stream>>>(xt, xb,  (int)n_xt);
        cvt_bf16_kernel<<<(int)(n_w  / 8 / 256), 256, 0, stream>>>(Wa, wab, (int)n_w);
        cvt_bf16_kernel<<<(int)(n_w  / 8 / 256), 256, 0, stream>>>(Wx, wxb, (int)n_w);

        // 256 row strips (64 rows) x 32 col strips; 8 waves/block share a col strip
        const int blocks = (NROWS / 64 / 8) * (DDIM / 32);   // 1024
        minlstm_bf16_kernel<<<blocks, 256, LDS_BYTES, stream>>>(
            xb, wab, wxb, xt, hprev, ba, bx, Lambda, out);
    } else {
        const int total_waves = (NROWS / 32) * (DDIM / 32);
        const int blocks      = total_waves * 32 / 256;      // 2048
        minlstm_f32src_kernel<<<blocks, 256, 0, stream>>>(
            xt, hprev, Wa, Wx, ba, bx, Lambda, out);
    }
}
